// PowerSpectrogram_19954418057365
// MI455X (gfx1250) — matrix operand order repaired and checked by reading, not compiled, not run on hardware
//
#include <hip/hip_runtime.h>

typedef __attribute__((ext_vector_type(16))) _Float16 v16h;
typedef __attribute__((ext_vector_type(8)))  _Float16 v8h;
typedef __attribute__((ext_vector_type(8)))  float    v8f;

#define NBATCH   64
#define NSAMP    160000
#define WIN      160
#define HOP      80
#define TBINS    1999          // (160000-160)/80 + 1
#define NFREQ    257           // rfft bins of 512
#define NT_TILES 17            // ceil(257/16) freq tiles
#define KT_TILES 5             // 160/32 k-steps
#define MT       32            // frames per workgroup
#define FCHUNKS  63            // ceil(1999/32)
#define APAD     168           // LDS row stride in halves (336B = 21*16B, keeps b128 alignment)

// ---------------------------------------------------------------------------
// Basis init: pack DFT matrix (cos and -sin, each split into f16 hi + lo parts)
// as v_wmma_f32_16x16x32_f16 B fragments in this file's contiguous-k order (matches the A loads):
//   B element (kk within 0..31, col n within 0..15):
//     lane = (n & 15) + 16*(kk >= 16),  v16h slot j = kk & 15
// Layout: basis[((part*17 + ntile)*5 + ktile)*32 + lane] is one v16h (32B).
// part: 0 = cos_hi, 1 = -sin_hi, 2 = cos_lo, 3 = -sin_lo
// ---------------------------------------------------------------------------
__global__ __launch_bounds__(256) void basis_init_kernel(_Float16* __restrict__ basis) {
  const int gid = blockIdx.x * blockDim.x + threadIdx.x;
  const int total = 4 * NT_TILES * KT_TILES * 32;
  if (gid >= total) return;

  const int lane = gid & 31;
  const int kt   = (gid >> 5) % KT_TILES;
  const int nt   = (gid / (32 * KT_TILES)) % NT_TILES;
  const int part = gid / (32 * KT_TILES * NT_TILES);

  const int hi = lane >> 4;
  const int c  = nt * 16 + (lane & 15);       // frequency bin

  v16h vals;
#pragma unroll
  for (int j = 0; j < 16; ++j) {
    const int w = kt * 32 + hi * 16 + j;      // window sample index, 0..159
    _Float16 o = (_Float16)0.0f;
    if (c < NFREQ) {
      const int ph = (w * c) & 511;           // exact mod-512 phase
      const float th = (float)ph * (3.14159265358979323846f / 256.0f);
      const float base = (part & 1) ? -__sinf(th) : __cosf(th);
      const _Float16 h = (_Float16)base;
      o = (part < 2) ? h : (_Float16)(base - (float)h);  // hi or residual lo
    }
    vals[j] = o;
  }
  *(v16h*)(basis + (size_t)gid * 16) = vals;  // 32B contiguous store per thread
}

// ---------------------------------------------------------------------------
// Main kernel: one workgroup = 32 frames x all 257 bins (cos & sin GEMMs),
// accumulated with v_wmma_f32_16x16x32_f16 using hi/lo split operands:
//   acc += Al*Bh + Ah*Bl + Ah*Bh   (~f32 accuracy)
// ---------------------------------------------------------------------------
__global__ __launch_bounds__(256) void spec_wmma_kernel(const float* __restrict__ audio,
                                                        const _Float16* __restrict__ basis,
                                                        float* __restrict__ out) {
  __shared__ __attribute__((aligned(16))) _Float16 sAh[MT][APAD];
  __shared__ __attribute__((aligned(16))) _Float16 sAl[MT][APAD];

  const int b   = blockIdx.y;
  const int f0  = blockIdx.x * MT;
  const int tid = threadIdx.x;

  // Stage 32 overlapping frames into LDS, split into f16 hi + residual lo.
  for (int idx = tid; idx < MT * WIN; idx += 256) {
    const int m = idx / WIN;
    const int w = idx - m * WIN;
    const int f = f0 + m;
    float v = 0.0f;
    if (f < TBINS) v = audio[(size_t)b * NSAMP + (size_t)f * HOP + w];
    const _Float16 h = (_Float16)v;
    sAh[m][w] = h;
    sAl[m][w] = (_Float16)(v - (float)h);
  }
  __syncthreads();

  const int wave = tid >> 5;
  const int lane = tid & 31;
  const int hi   = lane >> 4;
  const v16h* __restrict__ Bp = (const v16h*)basis;

  // 2 M-tiles x 17 N-tiles = 34 output tile-pairs across 8 waves.
  for (int p = wave; p < 2 * NT_TILES; p += 8) {
    const int mt = p / NT_TILES;
    const int nt = p - mt * NT_TILES;

    // A-fragment addressing (16-bit 16x32 A layout): lane holds row M=lane&15;
    // its 16 halves are one contiguous run [kb, kb+16), kb = 16*(lane>=16) (matches B packing).
    const int mrow  = mt * 16 + (lane & 15);
    const int kbase = hi * 16;

    v8f accC = {};
    v8f accS = {};

#pragma unroll
    for (int kt = 0; kt < KT_TILES; ++kt) {
      const int off = kt * 32 + kbase;
      const v8h ah_lo = *(const v8h*)&sAh[mrow][off];
      const v8h ah_hi = *(const v8h*)&sAh[mrow][off + 8];
      const v8h al_lo = *(const v8h*)&sAl[mrow][off];
      const v8h al_hi = *(const v8h*)&sAl[mrow][off + 8];
      const v16h ah = __builtin_shufflevector(ah_lo, ah_hi, 0,1,2,3,4,5,6,7,8,9,10,11,12,13,14,15);
      const v16h al = __builtin_shufflevector(al_lo, al_hi, 0,1,2,3,4,5,6,7,8,9,10,11,12,13,14,15);

      const int bbase = (nt * KT_TILES + kt) * 32 + lane;
      const v16h bCh = Bp[bbase + 0 * NT_TILES * KT_TILES * 32];
      const v16h bSh = Bp[bbase + 1 * NT_TILES * KT_TILES * 32];
      const v16h bCl = Bp[bbase + 2 * NT_TILES * KT_TILES * 32];
      const v16h bSl = Bp[bbase + 3 * NT_TILES * KT_TILES * 32];

      accC = __builtin_amdgcn_wmma_f32_16x16x32_f16(false, al, false, bCh, (short)0, accC, false, false);
      accC = __builtin_amdgcn_wmma_f32_16x16x32_f16(false, ah, false, bCl, (short)0, accC, false, false);
      accC = __builtin_amdgcn_wmma_f32_16x16x32_f16(false, ah, false, bCh, (short)0, accC, false, false);
      accS = __builtin_amdgcn_wmma_f32_16x16x32_f16(false, al, false, bSh, (short)0, accS, false, false);
      accS = __builtin_amdgcn_wmma_f32_16x16x32_f16(false, ah, false, bSl, (short)0, accS, false, false);
      accS = __builtin_amdgcn_wmma_f32_16x16x32_f16(false, ah, false, bSh, (short)0, accS, false, false);
    }

    // D layout: lane holds N = lane&15; VGPR v holds M = v + 8*(lane>=16).
    const int k = nt * 16 + (lane & 15);
#pragma unroll
    for (int v = 0; v < 8; ++v) {
      const int f = f0 + mt * 16 + v + 8 * hi;
      if (f < TBINS && k < NFREQ) {
        const float re = accC[v];
        const float im = accS[v];
        out[((size_t)b * TBINS + f) * NFREQ + k] = (re * re + im * im) * (1.0f / 512.0f);
      }
    }
  }
}

extern "C" void kernel_launch(void* const* d_in, const int* in_sizes, int n_in,
                              void* d_out, int out_size, void* d_ws, size_t ws_size,
                              hipStream_t stream) {
  (void)in_sizes; (void)n_in; (void)out_size; (void)ws_size;
  const float* audio = (const float*)d_in[0];
  _Float16*    basis = (_Float16*)d_ws;           // 4*17*5*512 halves = 348,160 B
  float*       out   = (float*)d_out;

  const int basis_threads = 4 * NT_TILES * KT_TILES * 32;   // 10880
  basis_init_kernel<<<(basis_threads + 255) / 256, 256, 0, stream>>>(basis);

  dim3 grid(FCHUNKS, NBATCH);                      // 63 x 64 workgroups
  spec_wmma_kernel<<<grid, 256, 0, stream>>>(audio, basis, out);
}
